// MathFeatureMLP_70111046139935
// MI455X (gfx1250) — matrix operand order repaired and checked by reading, not compiled, not run on hardware
//
#include <hip/hip_runtime.h>
#include <hip/hip_bf16.h>

// ---------------------------------------------------------------------------
// MathFeatureMLP for MI455X (gfx1250, wave32, WMMA, async global->LDS)
//   feats[b] = [x(512) | P^T P (1024) | P P^T (256) | svd(16) | pad(16)] bf16
//   h1 = relu(feats @ W1 + b1)  -> WMMA bf16 GEMM, K=1824  (dominant, 60.7 GF)
//   h2 = relu(h1 @ W2 + b2)     -> WMMA bf16 GEMM, K=256
//   h3 = relu(h2 @ W3 + b3)     -> WMMA bf16 GEMM, K=128
//   out = h3 @ W4 + b4          -> f32 dot
// ---------------------------------------------------------------------------

typedef __attribute__((ext_vector_type(16))) __bf16 v16bf;
typedef __attribute__((ext_vector_type(8)))  float  v8f;

union FragBF { uint4 u[2]; v16bf v; };

#define BATCH   65536
#define KROWS   16
#define COLS    32
#define XLEN    512
#define FEAT    1808
#define KP1     1824      // FEAT padded to multiple of 32
#define N1      256
#define N2      128
#define N3      64
#define LDSW    40        // bf16 elems per LDS row (32 + 8 pad); 80B, 16B aligned

// ---- CDNA5 async global->LDS copy (ASYNCcnt path), per-lane 16B ----------
__device__ __forceinline__ void async_load_b128(const void* gptr, void* lptr) {
  unsigned lds_off = (unsigned)(uintptr_t)lptr;   // low 32 bits = LDS byte addr
  asm volatile("global_load_async_to_lds_b128 %0, %1, off"
               :: "v"(lds_off), "v"(gptr)
               : "memory");
}
__device__ __forceinline__ void wait_asynccnt0() {
  asm volatile("s_wait_asynccnt 0" ::: "memory");
}

// ---------------------------------------------------------------------------
// Weight convert + transpose: W[K][N] f32 -> Wt[N][Kp] bf16 (zero pad K->Kp)
// ---------------------------------------------------------------------------
__global__ void convert_transpose(const float* __restrict__ W,
                                  __hip_bfloat16* __restrict__ Wt,
                                  int Kdim, int Ndim, int Kp) {
  int tid = blockIdx.x * blockDim.x + threadIdx.x;
  if (tid >= Ndim * Kp) return;
  int n = tid / Kp;
  int k = tid - n * Kp;
  float v = (k < Kdim) ? W[(size_t)k * Ndim + n] : 0.0f;
  Wt[(size_t)n * Kp + k] = __float2bfloat16(v);
}

// ---------------------------------------------------------------------------
// Feature kernel: one wave32 per sample, 8 samples per 256-thread block.
// ---------------------------------------------------------------------------
__global__ __launch_bounds__(256) void feature_kernel(
    const float* __restrict__ x,          // [B][512]
    __hip_bfloat16* __restrict__ feats) { // [B][1824]
  __shared__ __align__(16) float Ps[8][XLEN];       // P, row-major [16][32]
  __shared__ float Gs[8][16 * 17];                  // Gram 16x16, padded

  const int tid  = threadIdx.x;
  const int lane = tid & 31;
  const int w    = tid >> 5;
  const int s    = blockIdx.x * 8 + w;

  const float* xs = x + (size_t)s * XLEN;
  __hip_bfloat16* f = feats + (size_t)s * KP1;

  // ---- load P into LDS and emit raw x features ----
  #pragma unroll
  for (int i = 0; i < 4; ++i) {
    float4 v = *(const float4*)(xs + lane * 16 + i * 4);
    *(float4*)(&Ps[w][lane * 16 + i * 4]) = v;
    f[lane * 16 + i * 4 + 0] = __float2bfloat16(v.x);
    f[lane * 16 + i * 4 + 1] = __float2bfloat16(v.y);
    f[lane * 16 + i * 4 + 2] = __float2bfloat16(v.z);
    f[lane * 16 + i * 4 + 3] = __float2bfloat16(v.w);
  }
  __syncthreads();

  // ---- gram_cols[i][j] = sum_k P[k][i]*P[k][j]; lane owns column j ----
  {
    const int j = lane;
    for (int i = 0; i < COLS; ++i) {
      float sum = 0.0f;
      #pragma unroll
      for (int k = 0; k < KROWS; ++k)
        sum += Ps[w][k * COLS + i] * Ps[w][k * COLS + j];
      f[XLEN + i * COLS + j] = __float2bfloat16(sum);
    }
  }

  // ---- gram_rows[i][j] = sum_c P[i][c]*P[j][c]; 8 entries per lane ----
  {
    const int gi = lane & 15;
    const int gj0 = (lane >> 4) * 8;
    for (int jj = 0; jj < 8; ++jj) {
      const int gj = gj0 + jj;
      float sum = 0.0f;
      #pragma unroll
      for (int c = 0; c < COLS; ++c)
        sum += Ps[w][gi * COLS + c] * Ps[w][gj * COLS + c];
      Gs[w][gi * 17 + gj] = sum;
      f[XLEN + 1024 + gi * 16 + gj] = __float2bfloat16(sum);
    }
  }
  __syncthreads();

  // ---- cyclic Jacobi eigensolve of 16x16 symmetric Gram (6 sweeps) ----
  for (int sweep = 0; sweep < 6; ++sweep) {
    for (int p = 0; p < 15; ++p) {
      for (int q = p + 1; q < 16; ++q) {
        const float app = Gs[w][p * 17 + p];
        const float aqq = Gs[w][q * 17 + q];
        const float apq = Gs[w][p * 17 + q];
        float c = 1.0f, sn = 0.0f;
        if (fabsf(apq) > 1e-12f) {           // uniform across the wave
          float tau = (aqq - app) / (2.0f * apq);
          float t = (tau >= 0.0f ? 1.0f : -1.0f) /
                    (fabsf(tau) + sqrtf(1.0f + tau * tau));
          c = rsqrtf(1.0f + t * t);
          sn = t * c;
        }
        if (lane < 16) {                      // row rotation, k = lane
          const int k = lane;
          float gpk = Gs[w][p * 17 + k], gqk = Gs[w][q * 17 + k];
          Gs[w][p * 17 + k] = c * gpk - sn * gqk;
          Gs[w][q * 17 + k] = sn * gpk + c * gqk;
        }
        __builtin_amdgcn_wave_barrier();
        if (lane < 16) {                      // column rotation
          const int k = lane;
          float gkp = Gs[w][k * 17 + p], gkq = Gs[w][k * 17 + q];
          Gs[w][k * 17 + p] = c * gkp - sn * gkq;
          Gs[w][k * 17 + q] = sn * gkp + c * gkq;
        }
        __builtin_amdgcn_wave_barrier();
      }
    }
  }

  // ---- singular values = sqrt(eig), sorted descending; zero the K pad ----
  if (lane < 16) {
    float ev = Gs[w][lane * 17 + lane];
    int rank = 0;
    for (int t = 0; t < 16; ++t) {
      float e2 = Gs[w][t * 17 + t];
      if (e2 > ev || (e2 == ev && t < lane)) rank++;
    }
    f[XLEN + 1024 + 256 + rank] = __float2bfloat16(sqrtf(fmaxf(ev, 0.0f)));
    f[FEAT + lane] = __float2bfloat16(0.0f);   // pad 1808..1823
  }
}

// ---------------------------------------------------------------------------
// WMMA bf16 GEMM + bias + ReLU:  C[M][N] = relu(A[M][Kp] @ Wt[N][Kp]^T + b)
// Block: 256 threads (8 wave32s), block tile 128x64, wave tile 32x32.
// Double-buffered LDS tiles filled with async global->LDS (ASYNCcnt).
// ---------------------------------------------------------------------------
__global__ __launch_bounds__(256) void gemm_bf16_relu(
    const __hip_bfloat16* __restrict__ A,    // [M][Kp] row-major
    const __hip_bfloat16* __restrict__ Wt,   // [N][Kp] row-major (W^T)
    const float* __restrict__ bias,          // [N]
    __hip_bfloat16* __restrict__ Cout,       // [M][N]
    int M, int Ntot, int Kp) {
  __shared__ __align__(16) __hip_bfloat16 Atile[2][128 * LDSW];
  __shared__ __align__(16) __hip_bfloat16 Btile[2][64 * LDSW];

  const int tid   = threadIdx.x;
  const int lane  = tid & 31;
  const int wid   = tid >> 5;
  const int m_w   = wid & 3;                 // 4 waves along M
  const int n_w   = wid >> 2;                // 2 waves along N
  const int mbase = blockIdx.y * 128;
  const int nbase = blockIdx.x * 64;

  // per-thread chunk coordinates for cooperative tile fills
  const int ar0 = (tid * 2) >> 2;            // A chunk 0 row
  const int ac0 = ((tid * 2) & 3) * 8;       // A chunk 0 col
  const int ar1 = (tid * 2 + 1) >> 2;        // A chunk 1 row
  const int ac1 = ((tid * 2 + 1) & 3) * 8;   // A chunk 1 col
  const int br  = tid >> 2;                  // B chunk row
  const int bc  = (tid & 3) * 8;             // B chunk col

  auto load_tiles = [&](int kt, int buf) {
    const int k0 = kt << 5;
    async_load_b128(A + (size_t)(mbase + ar0) * Kp + k0 + ac0,
                    &Atile[buf][ar0 * LDSW + ac0]);
    async_load_b128(A + (size_t)(mbase + ar1) * Kp + k0 + ac1,
                    &Atile[buf][ar1 * LDSW + ac1]);
    async_load_b128(Wt + (size_t)(nbase + br) * Kp + k0 + bc,
                    &Btile[buf][br * LDSW + bc]);
  };

  v8f acc[2][2];
  const v8f vzero = {0.f, 0.f, 0.f, 0.f, 0.f, 0.f, 0.f, 0.f};
  #pragma unroll
  for (int i = 0; i < 2; ++i)
    #pragma unroll
    for (int j = 0; j < 2; ++j) acc[i][j] = vzero;

  const int nK = Kp >> 5;

  // prologue: fill buffer 0
  load_tiles(0, 0);
  wait_asynccnt0();
  __syncthreads();

  for (int kt = 0; kt < nK; ++kt) {
    const int buf = kt & 1;
    // overlap: fetch next K-slice into the other buffer while we compute
    if (kt + 1 < nK) load_tiles(kt + 1, buf ^ 1);

    // fragments per ISA bf16 A (16x32) / B (32x16) lane layouts
    FragBF a[2], b[2];
    const int arow0 = m_w * 32 + (lane & 15);
    const int aoff  = (lane >> 4) * 8;
    #pragma unroll
    for (int i = 0; i < 2; ++i) {
      const int row = arow0 + i * 16;
      a[i].u[0] = *(const uint4*)(&Atile[buf][row * LDSW + aoff]);
      a[i].u[1] = *(const uint4*)(&Atile[buf][row * LDSW + aoff + 16]);
    }
    const int bcol0 = n_w * 32 + (lane & 15);
    const int bk    = (lane >> 4) * 8;
    #pragma unroll
    for (int j = 0; j < 2; ++j) {
      const int col = bcol0 + j * 16;
      b[j].u[0] = *(const uint4*)(&Btile[buf][col * LDSW + bk]);
      b[j].u[1] = *(const uint4*)(&Btile[buf][col * LDSW + bk + 16]);
    }

    #pragma unroll
    for (int i = 0; i < 2; ++i)
      #pragma unroll
      for (int j = 0; j < 2; ++j)
        acc[i][j] = __builtin_amdgcn_wmma_f32_16x16x32_bf16(
            false, a[i].v, false, b[j].v, (short)0, acc[i][j], false, false);

    // next tile fully in LDS + all waves done reading this buffer
    wait_asynccnt0();
    __syncthreads();
  }

  // epilogue: bias + ReLU, store bf16 per C/D lane layout
  #pragma unroll
  for (int j = 0; j < 2; ++j) {
    const int col = nbase + n_w * 32 + j * 16 + (lane & 15);
    const float bv = bias[col];
    #pragma unroll
    for (int i = 0; i < 2; ++i) {
      const int rbase = mbase + m_w * 32 + i * 16 + ((lane >> 4) * 8);
      #pragma unroll
      for (int r = 0; r < 8; ++r) {
        float v = acc[i][j][r] + bv;
        v = v > 0.0f ? v : 0.0f;
        Cout[(size_t)(rbase + r) * Ntot + col] = __float2bfloat16(v);
      }
    }
  }
}

// ---------------------------------------------------------------------------
// Final layer: out[b] = dot(h3[b], W4) + b4   (64-wide, f32 result)
// ---------------------------------------------------------------------------
__global__ __launch_bounds__(256) void final_layer(
    const __hip_bfloat16* __restrict__ h3,   // [B][64]
    const float* __restrict__ W4,            // [64]
    const float* __restrict__ b4,            // [1]
    float* __restrict__ out) {
  const int r = blockIdx.x * blockDim.x + threadIdx.x;
  float acc = b4[0];
  const __hip_bfloat16* hr = h3 + (size_t)r * 64;
  #pragma unroll
  for (int j = 0; j < 64; ++j)
    acc += __bfloat162float(hr[j]) * W4[j];
  out[r] = acc;
}

// ---------------------------------------------------------------------------
// Launch
// ---------------------------------------------------------------------------
extern "C" void kernel_launch(void* const* d_in, const int* in_sizes, int n_in,
                              void* d_out, int out_size, void* d_ws, size_t ws_size,
                              hipStream_t stream) {
  (void)in_sizes; (void)n_in; (void)out_size; (void)ws_size;

  const float* x  = (const float*)d_in[0];
  const float* W1 = (const float*)d_in[1];
  const float* b1 = (const float*)d_in[2];
  const float* W2 = (const float*)d_in[3];
  const float* b2 = (const float*)d_in[4];
  const float* W3 = (const float*)d_in[5];
  const float* b3 = (const float*)d_in[6];
  const float* W4 = (const float*)d_in[7];
  const float* b4 = (const float*)d_in[8];
  float* out = (float*)d_out;

  // workspace layout (bytes, all 256-aligned)
  char* base = (char*)d_ws;
  size_t off = 0;
  __hip_bfloat16* W1t = (__hip_bfloat16*)(base + off); off += (size_t)N1 * KP1 * 2;   // 933,888
  __hip_bfloat16* W2t = (__hip_bfloat16*)(base + off); off += (size_t)N2 * N1 * 2;    //  65,536
  __hip_bfloat16* W3t = (__hip_bfloat16*)(base + off); off += (size_t)N3 * N2 * 2;    //  16,384
  __hip_bfloat16* feats = (__hip_bfloat16*)(base + off); off += (size_t)BATCH * KP1 * 2;
  __hip_bfloat16* h1  = (__hip_bfloat16*)(base + off); off += (size_t)BATCH * N1 * 2;
  __hip_bfloat16* h2  = (__hip_bfloat16*)(base + off); off += (size_t)BATCH * N2 * 2;
  __hip_bfloat16* h3  = (__hip_bfloat16*)(base + off); off += (size_t)BATCH * N3 * 2;

  // weights -> bf16, transposed [N][Kp]
  {
    int c1 = N1 * KP1, c2 = N2 * N1, c3 = N3 * N2;
    convert_transpose<<<(c1 + 255) / 256, 256, 0, stream>>>(W1, W1t, FEAT, N1, KP1);
    convert_transpose<<<(c2 + 255) / 256, 256, 0, stream>>>(W2, W2t, N1, N2, N1);
    convert_transpose<<<(c3 + 255) / 256, 256, 0, stream>>>(W3, W3t, N2, N3, N2);
  }

  // features (grams + Jacobi singular values), bf16, K-padded
  feature_kernel<<<BATCH / 8, 256, 0, stream>>>(x, feats);

  // layer 1: [65536 x 1824] @ [1824 x 256]
  gemm_bf16_relu<<<dim3(N1 / 64, BATCH / 128), 256, 0, stream>>>(
      feats, W1t, b1, h1, BATCH, N1, KP1);
  // layer 2: [65536 x 256] @ [256 x 128]
  gemm_bf16_relu<<<dim3(N2 / 64, BATCH / 128), 256, 0, stream>>>(
      h1, W2t, b2, h2, BATCH, N2, N1);
  // layer 3: [65536 x 128] @ [128 x 64]
  gemm_bf16_relu<<<dim3(N3 / 64, BATCH / 128), 256, 0, stream>>>(
      h2, W3t, b3, h3, BATCH, N3, N2);
  // layer 4: dot-64 + bias
  final_layer<<<BATCH / 256, 256, 0, stream>>>(h3, W4, b4, out);
}
